// ExpertTokenMLPLoRABlock_41764261987094
// MI455X (gfx1250) — matrix operand order repaired and checked by reading, not compiled, not run on hardware
//
#include <hip/hip_runtime.h>

// ---------------------------------------------------------------------------
// Transformer block for MI455X (gfx1250, wave32, WMMA).
// All large matmuls use v_wmma_f32_16x16x32_bf16 (f32 accum, bf16 operands).
// Attention is flash-style: scores never hit HBM. Softmax row reductions use
// DPP8/DPP16 (pure VALU) instead of LDS bpermute round-trips.
// GEMM: 128x64 block tile, 4 waves, wave = 32x64 (2x4 WMMA register tile),
//       double-buffered LDS with register-staged global prefetch,
//       epilogue specialized by template (no runtime branches).
// ---------------------------------------------------------------------------

#define B_   8
#define S_   1024
#define D_   768
#define H_   12
#define HD_  64
#define DFF_ 3072
#define BS_  (B_ * S_)

typedef unsigned short bf16_t;
typedef __attribute__((ext_vector_type(16))) __bf16       v16bf;
typedef __attribute__((ext_vector_type(8)))  float        v8f;
typedef __attribute__((ext_vector_type(4)))  unsigned int u32x4;

union FragU { u32x4 u[2]; v16bf v; };  // 32 bytes = 8 VGPRs = one WMMA operand

__device__ __forceinline__ bf16_t f2bf(float f) {
  union { float f; unsigned u; } c; c.f = f;
  unsigned r = c.u + 0x7FFFu + ((c.u >> 16) & 1u);   // RNE
  return (bf16_t)(r >> 16);
}
__device__ __forceinline__ float bf2f(bf16_t h) {
  union { unsigned u; float f; } c; c.u = ((unsigned)h) << 16;
  return c.f;
}

__device__ __forceinline__ v8f wmma_bf16(v16bf a, v16bf b, v8f c) {
  // (neg_a, A, neg_b, B, c_mod, C, reuse_a, reuse_b)
  return __builtin_amdgcn_wmma_f32_16x16x32_bf16(false, a, false, b,
                                                 (short)0, c, false, false);
}

// --- 16-lane reductions in pure VALU via DPP (no LDS unit round trips) -----
// xor-1/2/4 inside 8-lane groups: DPP8 arbitrary permute.
template <int M>
__device__ __forceinline__ float dpp8_xor(float x) {
  constexpr int sel = (0 ^ M) | ((1 ^ M) << 3) | ((2 ^ M) << 6) |
                      ((3 ^ M) << 9) | ((4 ^ M) << 12) | ((5 ^ M) << 15) |
                      ((6 ^ M) << 18) | ((7 ^ M) << 21);
  return __builtin_bit_cast(
      float, __builtin_amdgcn_mov_dpp8(__builtin_bit_cast(int, x), sel));
}
// lane ^ 8 within each 16-lane row == DPP16 row_ror:8 (ctrl 0x128).
__device__ __forceinline__ float dpp16_xor8(float x) {
  return __builtin_bit_cast(
      float, __builtin_amdgcn_update_dpp(0, __builtin_bit_cast(int, x),
                                         0x128, 0xf, 0xf, true));
}
__device__ __forceinline__ float red16_max(float v) {
  v = fmaxf(v, dpp8_xor<1>(v));
  v = fmaxf(v, dpp8_xor<2>(v));
  v = fmaxf(v, dpp8_xor<4>(v));
  v = fmaxf(v, dpp16_xor8(v));
  return v;
}
__device__ __forceinline__ float red16_sum(float v) {
  v += dpp8_xor<1>(v);
  v += dpp8_xor<2>(v);
  v += dpp8_xor<4>(v);
  v += dpp16_xor8(v);
  return v;
}

// ---------------------------------------------------------------------------
// LayerNorm: one block per token (D=768 = 256*3), writes bf16 activations.
// ---------------------------------------------------------------------------
__global__ void __launch_bounds__(256)
ln_k(const float* __restrict__ X, const float* __restrict__ g,
     const float* __restrict__ bta, bf16_t* __restrict__ Y)
{
  __shared__ float sh[512];
  const int row = blockIdx.x, t = threadIdx.x;
  const float* xr = &X[(size_t)row * D_];
  float x0 = xr[t], x1 = xr[t + 256], x2 = xr[t + 512];
  sh[t]       = x0 + x1 + x2;
  sh[256 + t] = x0 * x0 + x1 * x1 + x2 * x2;
  __syncthreads();
  for (int o = 128; o > 0; o >>= 1) {
    if (t < o) { sh[t] += sh[t + o]; sh[256 + t] += sh[256 + t + o]; }
    __syncthreads();
  }
  float mu   = sh[0] * (1.f / D_);
  float var  = sh[256] * (1.f / D_) - mu * mu;
  float rstd = rsqrtf(var + 1e-6f);
  bf16_t* yr = &Y[(size_t)row * D_];
  yr[t]       = f2bf((x0 - mu) * rstd * g[t]       + bta[t]);
  yr[t + 256] = f2bf((x1 - mu) * rstd * g[t + 256] + bta[t + 256]);
  yr[t + 512] = f2bf((x2 - mu) * rstd * g[t + 512] + bta[t + 512]);
}

// ---------------------------------------------------------------------------
// WMMA GEMM: C[M,N] = A_bf16[M,K] x W_f32[K,N] + bias (+res).
// Double-buffered LDS; global prefetch issued before the WMMA burst so VMEM
// latency hides under matrix ops; one barrier per 32-wide k-step.
// ---------------------------------------------------------------------------
template <bool HAS_RES, bool OUT_F, bool OUT_B>
__global__ void __launch_bounds__(128)
gemm128(const bf16_t* __restrict__ A, const float* __restrict__ W,
        const float* __restrict__ bias, const float* __restrict__ res,
        float* __restrict__ Cf, bf16_t* __restrict__ Cbf,
        int M, int N, int K)
{
  __shared__ bf16_t As[2][128 * 40];  // [row][k], stride 40 halves (16B-align)
  __shared__ bf16_t Ws[2][64 * 40];   // [n][k] column-major (WMMA B layout)
  const int tid  = threadIdx.x;
  const int wave = tid >> 5, lane = tid & 31;
  const int hi   = lane >> 4, ln16 = lane & 15;
  const int mBase = blockIdx.y * 128, nBase = blockIdx.x * 64;

  v8f acc[2][4];
#pragma unroll
  for (int mi = 0; mi < 2; ++mi)
#pragma unroll
    for (int nt = 0; nt < 4; ++nt)
#pragma unroll
      for (int i = 0; i < 8; ++i) acc[mi][nt][i] = 0.f;

  // A staging: thread owns 4x 16B chunks, rows tid>>2 + {0,32,64,96}.
  const int aRow = tid >> 2, aCc = tid & 3;
  const bf16_t* aSrc = A + (size_t)(mBase + aRow) * K + aCc * 8;
  const size_t  aRowStride = (size_t)32 * K;
  // W staging: thread owns k-pair kkp (2 rows) x 8 n-values; packed b32 stores.
  const int kkp = tid >> 3, ng = tid & 7;
  const float* wSrc = W + (size_t)(2 * kkp) * N + nBase + ng * 8;

  u32x4    aReg[4];
  unsigned wReg[8];

  auto globalFetch = [&](int k0) {
#pragma unroll
    for (int i = 0; i < 4; ++i)
      aReg[i] = *(const u32x4*)(aSrc + i * aRowStride + k0);
    const float* w0 = wSrc + (size_t)k0 * N;
    const float* w1 = w0 + N;
#pragma unroll
    for (int i = 0; i < 8; ++i)
      wReg[i] = (unsigned)f2bf(w0[i]) | ((unsigned)f2bf(w1[i]) << 16);
  };
  auto ldsStore = [&](int buf) {
    bf16_t* ad = &As[buf][aRow * 40 + aCc * 8];
#pragma unroll
    for (int i = 0; i < 4; ++i)
      *(u32x4*)(ad + i * 32 * 40) = aReg[i];
#pragma unroll
    for (int i = 0; i < 8; ++i)
      *(unsigned*)&Ws[buf][(ng * 8 + i) * 40 + 2 * kkp] = wReg[i];
  };

  globalFetch(0);
  ldsStore(0);
  int cur = 0;

  for (int k0 = 0; k0 < K; k0 += 32) {
    __syncthreads();                       // tiles in As/Ws[cur] are ready
    const bool more = (k0 + 32) < K;
    if (more) globalFetch(k0 + 32);        // overlap VMEM with WMMA below

    FragU fa[2], fb[4];
#pragma unroll
    for (int mi = 0; mi < 2; ++mi) {
      const bf16_t* ap = &As[cur][(wave * 32 + mi * 16 + ln16) * 40];
      fa[mi].u[0] = *(const u32x4*)&ap[hi * 16];
      fa[mi].u[1] = *(const u32x4*)&ap[hi * 16 + 8];
    }
#pragma unroll
    for (int nt = 0; nt < 4; ++nt) {
      const bf16_t* bp = &Ws[cur][(nt * 16 + ln16) * 40 + hi * 16];
      fb[nt].u[0] = *(const u32x4*)&bp[0];
      fb[nt].u[1] = *(const u32x4*)&bp[8];
    }
#pragma unroll
    for (int mi = 0; mi < 2; ++mi)
#pragma unroll
      for (int nt = 0; nt < 4; ++nt)
        acc[mi][nt] = wmma_bf16(fa[mi].v, fb[nt].v, acc[mi][nt]);

    if (more) ldsStore(cur ^ 1);           // safe: buffer last read in previous iter, before the barrier
    cur ^= 1;
  }

  // Branch-free specialized epilogue. C/D layout: row = r + 8*hi, col = ln16.
#pragma unroll
  for (int mi = 0; mi < 2; ++mi)
#pragma unroll
    for (int nt = 0; nt < 4; ++nt) {
      int n = nBase + nt * 16 + ln16;
      float bv = bias[n];
#pragma unroll
      for (int r = 0; r < 8; ++r) {
        int m = mBase + wave * 32 + mi * 16 + r + hi * 8;
        size_t off = (size_t)m * N + n;
        float v = acc[mi][nt][r] + bv;
        if constexpr (HAS_RES) v += res[off];
        if constexpr (OUT_F)   Cf[off]  = v;
        if constexpr (OUT_B)   Cbf[off] = f2bf(v);
      }
    }
}

// ---------------------------------------------------------------------------
// Flash attention: one block = (b, h, 64 q-rows); 4 waves x 16 q-rows each.
// k-tiles of 64; Q.K^T and P.V via WMMA; online softmax in registers.
// ---------------------------------------------------------------------------
__global__ void __launch_bounds__(128)
flash_attn(const bf16_t* __restrict__ Q, const bf16_t* __restrict__ Km,
           const bf16_t* __restrict__ V, const float* __restrict__ mask,
           bf16_t* __restrict__ O)
{
  __shared__ bf16_t VT[64 * 72];       // [d][key] (B layout for P.V)
  __shared__ bf16_t Pt[4 * 16 * 72];   // per-wave P tile [q][key]
  const int tid  = threadIdx.x;
  const int wave = tid >> 5, lane = tid & 31;
  const int hi   = lane >> 4, ln16 = lane & 15;
  const int qBase = blockIdx.x * 64;
  const int b = blockIdx.y / H_, h = blockIdx.y % H_;
  const size_t headOff = (size_t)h * HD_;
  const int qrow0 = qBase + wave * 16;

  v8f outacc[4];
  float mrow[8], lrow[8];
  for (int t = 0; t < 4; ++t) for (int i = 0; i < 8; ++i) outacc[t][i] = 0.f;
  for (int r = 0; r < 8; ++r) { mrow[r] = -1e30f; lrow[r] = 0.f; }

  for (int kt = 0; kt < S_ / 64; ++kt) {
    __syncthreads();
    // Stage V tile transposed VT[d][key]: 4x b128 loads + register unpack.
    {
      int key = tid >> 1, dh = tid & 1;
      const u32x4* vp = (const u32x4*)
          &V[((size_t)(b * S_) + kt * 64 + key) * D_ + headOff + dh * 32];
#pragma unroll
      for (int c = 0; c < 4; ++c) {
        u32x4 w = vp[c];
#pragma unroll
        for (int j = 0; j < 4; ++j) {
          unsigned word = w[j];
          int d = dh * 32 + c * 8 + j * 2;
          VT[d * 72 + key]       = (bf16_t)(word & 0xffffu);
          VT[(d + 1) * 72 + key] = (bf16_t)(word >> 16);
        }
      }
    }
    __syncthreads();

    // scores(16q x 64k) = Q x K^T, contraction over HD=64 (2 WMMA k-steps).
    v8f sc[4];
    for (int t = 0; t < 4; ++t) for (int i = 0; i < 8; ++i) sc[t][i] = 0.f;
#pragma unroll
    for (int ds0 = 0; ds0 < HD_; ds0 += 32) {
      FragU fq, fk[4];
      {
        const bf16_t* qp =
            &Q[((size_t)(b * S_) + qrow0 + ln16) * D_ + headOff + ds0];
        fq.u[0] = *(const u32x4*)&qp[hi * 16];
        fq.u[1] = *(const u32x4*)&qp[hi * 16 + 8];
      }
#pragma unroll
      for (int nt = 0; nt < 4; ++nt) {
        int key = kt * 64 + nt * 16 + ln16;
        const bf16_t* kp =
            &Km[((size_t)(b * S_) + key) * D_ + headOff + ds0 + hi * 16];
        fk[nt].u[0] = *(const u32x4*)&kp[0];
        fk[nt].u[1] = *(const u32x4*)&kp[8];
      }
#pragma unroll
      for (int nt = 0; nt < 4; ++nt)
        sc[nt] = wmma_bf16(fq.v, fk[nt].v, sc[nt]);
    }

    // scale + additive mask + online softmax (per C/D row); DPP reductions.
#pragma unroll
    for (int r = 0; r < 8; ++r) {
      int q = qrow0 + r + hi * 8;
      float rm = -1e30f;
#pragma unroll
      for (int nt = 0; nt < 4; ++nt) {
        int kcol = kt * 64 + nt * 16 + ln16;
        float s = sc[nt][r] * 0.125f + mask[(size_t)q * S_ + kcol];
        sc[nt][r] = s;
        rm = fmaxf(rm, s);
      }
      rm = red16_max(rm);
      float mn   = fmaxf(mrow[r], rm);
      float corr = __expf(mrow[r] - mn);
      mrow[r] = mn;
      float rs = 0.f;
#pragma unroll
      for (int nt = 0; nt < 4; ++nt) {
        float pv = __expf(sc[nt][r] - mn);
        sc[nt][r] = pv;
        rs += pv;
      }
      rs = red16_sum(rs);
      lrow[r] = lrow[r] * corr + rs;
#pragma unroll
      for (int nt = 0; nt < 4; ++nt) outacc[nt][r] *= corr;
    }

    // Repack P (C/D layout) -> LDS (A layout source); per-wave private region,
    // LDS is in-order within a wave.
    bf16_t* pw = &Pt[wave * 16 * 72];
#pragma unroll
    for (int nt = 0; nt < 4; ++nt)
#pragma unroll
      for (int r = 0; r < 8; ++r)
        pw[(r + hi * 8) * 72 + nt * 16 + ln16] = f2bf(sc[nt][r]);

    // out += P x V, contraction over 64 keys (2 WMMA k-steps).
#pragma unroll
    for (int ks = 0; ks < 64; ks += 32) {
      FragU fp, fv[4];
      const bf16_t* pr = &Pt[wave * 16 * 72 + ln16 * 72 + ks];
      fp.u[0] = *(const u32x4*)&pr[hi * 16];
      fp.u[1] = *(const u32x4*)&pr[hi * 16 + 8];
#pragma unroll
      for (int nt = 0; nt < 4; ++nt) {
        const bf16_t* vp2 = &VT[(nt * 16 + ln16) * 72 + ks + hi * 16];
        fv[nt].u[0] = *(const u32x4*)&vp2[0];
        fv[nt].u[1] = *(const u32x4*)&vp2[8];
      }
#pragma unroll
      for (int nt = 0; nt < 4; ++nt)
        outacc[nt] = wmma_bf16(fp.v, fv[nt].v, outacc[nt]);
    }
  }

#pragma unroll
  for (int nt = 0; nt < 4; ++nt)
#pragma unroll
    for (int r = 0; r < 8; ++r) {
      int q = qrow0 + r + hi * 8;
      float v = outacc[nt][r] / lrow[r];
      O[((size_t)(b * S_) + q) * D_ + headOff + nt * 16 + ln16] = f2bf(v);
    }
}

// ---------------------------------------------------------------------------
// LoRA rank-16 down-projection: T[row,16] = X_bf16[row,:] @ Am[:,16]
// (only for expert tokens, gated on device-side *nbt).
// ---------------------------------------------------------------------------
__global__ void __launch_bounds__(256)
lora_down(const bf16_t* __restrict__ X, const float* __restrict__ Am,
          const int* __restrict__ nbt, float* __restrict__ T, int Kdim)
{
  int id = blockIdx.x * 256 + threadIdx.x;
  int row = id >> 4, j = id & 15;
  if (row >= BS_) return;
  if ((row & (S_ - 1)) < *nbt) return;
  const bf16_t* xr = &X[(size_t)row * Kdim];
  float acc = 0.f;
  for (int k = 0; k < Kdim; ++k)
    acc += bf2f(xr[k]) * Am[(size_t)k * 16 + j];
  T[(size_t)row * 16 + j] = acc;
}

// LoRA up-projection add: Out[row,n] += T[row,:16] @ Bm[:16,n] (expert rows).
__global__ void __launch_bounds__(256)
lora_up(const float* __restrict__ T, const float* __restrict__ Bm,
        const int* __restrict__ nbt, float* __restrict__ Out, int Ndim)
{
  long id = (long)blockIdx.x * 256 + threadIdx.x;
  if (id >= (long)BS_ * Ndim) return;
  int row = (int)(id / Ndim), n = (int)(id % Ndim);
  if ((row & (S_ - 1)) < *nbt) return;
  float acc = 0.f;
#pragma unroll
  for (int j = 0; j < 16; ++j)
    acc += T[(size_t)row * 16 + j] * Bm[(size_t)j * Ndim + n];
  Out[id] += acc;
}

// Exact GELU, f32 in -> bf16 out (feeds the W2 WMMA GEMM).
__global__ void __launch_bounds__(256)
gelu_k(const float* __restrict__ Hf, bf16_t* __restrict__ Hb, long n)
{
  long id = (long)blockIdx.x * 256 + threadIdx.x;
  if (id >= n) return;
  float x = Hf[id];
  Hb[id] = f2bf(0.5f * x * (1.f + erff(x * 0.70710678118654752f)));
}

// ---------------------------------------------------------------------------
extern "C" void kernel_launch(void* const* d_in, const int* in_sizes, int n_in,
                              void* d_out, int out_size, void* d_ws, size_t ws_size,
                              hipStream_t stream)
{
  const float* x    = (const float*)d_in[0];
  const float* mask = (const float*)d_in[1];
  const int*   nbt  = (const int*)  d_in[2];
  const float* ln1g = (const float*)d_in[3];
  const float* ln1b = (const float*)d_in[4];
  const float* wq   = (const float*)d_in[5];
  const float* bq   = (const float*)d_in[6];
  const float* wk   = (const float*)d_in[7];
  const float* bk   = (const float*)d_in[8];
  const float* wv   = (const float*)d_in[9];
  const float* bv   = (const float*)d_in[10];
  const float* wo   = (const float*)d_in[11];
  const float* bo   = (const float*)d_in[12];
  const float* ln2g = (const float*)d_in[13];
  const float* ln2b = (const float*)d_in[14];
  const float* w1   = (const float*)d_in[15];
  const float* b1   = (const float*)d_in[16];
  const float* w2   = (const float*)d_in[17];
  const float* b2   = (const float*)d_in[18];
  const float* la1  = (const float*)d_in[19];
  const float* lb1  = (const float*)d_in[20];
  const float* la2  = (const float*)d_in[21];
  const float* lb2  = (const float*)d_in[22];
  float* out = (float*)d_out;
  (void)in_sizes; (void)n_in; (void)out_size; (void)ws_size;

  char* p = (char*)d_ws;
  auto alloc = [&](size_t bytes) {
    char* r = p; p += (bytes + 255) & ~((size_t)255); return r;
  };
  bf16_t* h_bf   = (bf16_t*)alloc((size_t)BS_ * D_   * 2);
  bf16_t* q_bf   = (bf16_t*)alloc((size_t)BS_ * D_   * 2);
  bf16_t* k_bf   = (bf16_t*)alloc((size_t)BS_ * D_   * 2);
  bf16_t* v_bf   = (bf16_t*)alloc((size_t)BS_ * D_   * 2);
  bf16_t* a_bf   = (bf16_t*)alloc((size_t)BS_ * D_   * 2);
  float*  x1_f   = (float*) alloc((size_t)BS_ * D_   * 4);
  bf16_t* h2_bf  = (bf16_t*)alloc((size_t)BS_ * D_   * 2);
  float*  hid_f  = (float*) alloc((size_t)BS_ * DFF_ * 4);
  bf16_t* hid_bf = (bf16_t*)alloc((size_t)BS_ * DFF_ * 2);
  float*  t1     = (float*) alloc((size_t)BS_ * 16   * 4);
  float*  t2     = (float*) alloc((size_t)BS_ * 16   * 4);

  dim3 blk128(128), blk256(256);
  dim3 gD(D_ / 64, BS_ / 128);     // 12 x 64 tiles
  dim3 gF(DFF_ / 64, BS_ / 128);   // 48 x 64 tiles

  // --- attention branch ---
  ln_k<<<BS_, blk256, 0, stream>>>(x, ln1g, ln1b, h_bf);
  gemm128<false, false, true><<<gD, blk128, 0, stream>>>(
      h_bf, wq, bq, nullptr, nullptr, q_bf, BS_, D_, D_);
  gemm128<false, false, true><<<gD, blk128, 0, stream>>>(
      h_bf, wk, bk, nullptr, nullptr, k_bf, BS_, D_, D_);
  gemm128<false, false, true><<<gD, blk128, 0, stream>>>(
      h_bf, wv, bv, nullptr, nullptr, v_bf, BS_, D_, D_);
  flash_attn<<<dim3(S_ / 64, B_ * H_), blk128, 0, stream>>>(
      q_bf, k_bf, v_bf, mask, a_bf);
  gemm128<true, true, false><<<gD, blk128, 0, stream>>>(
      a_bf, wo, bo, x, x1_f, nullptr, BS_, D_, D_);

  // --- MLP branch with token-range LoRA ---
  ln_k<<<BS_, blk256, 0, stream>>>(x1_f, ln2g, ln2b, h2_bf);
  gemm128<false, true, false><<<gF, blk128, 0, stream>>>(
      h2_bf, w1, b1, nullptr, hid_f, nullptr, BS_, DFF_, D_);
  lora_down<<<(BS_ * 16) / 256, blk256, 0, stream>>>(h2_bf, la1, nbt, t1, D_);
  lora_up<<<(unsigned)(((long)BS_ * DFF_) / 256), blk256, 0, stream>>>(
      t1, lb1, nbt, hid_f, DFF_);
  gelu_k<<<(unsigned)(((long)BS_ * DFF_ + 255) / 256), blk256, 0, stream>>>(
      hid_f, hid_bf, (long)BS_ * DFF_);
  gemm128<true, true, false><<<gD, blk128, 0, stream>>>(
      hid_bf, w2, b2, x1_f, out, nullptr, BS_, D_, DFF_);
  lora_down<<<(BS_ * 16) / 256, blk256, 0, stream>>>(hid_bf, la2, nbt, t2, DFF_);
  lora_up<<<(unsigned)(((long)BS_ * D_) / 256), blk256, 0, stream>>>(
      t2, lb2, nbt, out, D_);
}
